// TensorProductMultiLayerPerceptron_10118942949798
// MI455X (gfx1250) — matrix operand order repaired and checked by reading, not compiled, not run on hardware
//
#include <hip/hip_runtime.h>

typedef __bf16 bf16_t;
typedef __attribute__((ext_vector_type(16))) __bf16 v16bf;
typedef __attribute__((ext_vector_type(8)))  __bf16 v8bf;
typedef __attribute__((ext_vector_type(8)))  float  v8f;

#define D_EMB   8
#define NW      64
#define MUL0    128
#define MUL1    64
#define MTILE   128
#define X1S     320   // MUL0 + 3*MUL1
#define OUTS    960   // 128 + 384 + 192 + 64 + 192

// LDS layout (bf16 element offsets), all matrices stored TRANSPOSED [N][K]
#define OFF_W0T 0        // [64][8]
#define OFF_W1T 512      // [64][64]
#define OFF_W2T 4608     // [64][64]
#define OFF_PT  8704     // [448][64]  (p000|p011|p102|p113|p114 along N)
#define OFF_H   37376    // [128][64] activation strip buffer
#define LDS_ELEMS 45568  // 91,136 bytes

__device__ __forceinline__ v16bf concat16(v8bf lo, v8bf hi) {
  return __builtin_shufflevector(lo, hi, 0,1,2,3,4,5,6,7,8,9,10,11,12,13,14,15);
}

__device__ __forceinline__ v8f wmma_bf16(v16bf a, v16bf b, v8f c) {
  return __builtin_amdgcn_wmma_f32_16x16x32_bf16(false, a, false, b, (short)0, c, false, false);
}

// Branchless tanh-gelu on raw HW transcendentals:
// tanh(y) = sign(y) * (1 - e) / (1 + e), e = exp(-2|y|) = exp2(-2*log2(e)*|y|)
__device__ __forceinline__ float gelu_fast(float x) {
  float y = 0.7978845608028654f * (x + 0.044715f * x * x * x);
  float e = __builtin_amdgcn_exp2f(-2.8853900817779268f * __builtin_fabsf(y));
  float t = (1.0f - e) * __builtin_amdgcn_rcpf(1.0f + e);
  t = __builtin_copysignf(t, y);
  return 0.5f * x * (1.0f + t);
}

// A fragment 16x32 bf16 from row-major h[row][k] (ld = NW).
// Lanes 0-15 row M=lane hold K {0..7,16..23}; lanes 16-31 hold K {8..15,24..31}.
__device__ __forceinline__ v16bf load_a_frag(const bf16_t* h, int m0, int lane, int kbase) {
  const int row  = m0 + (lane & 15);
  const int koff = (lane >> 4) << 3;                 // 0 or 8
  const bf16_t* p = h + row * NW + kbase + koff;
  v8bf lo = *(const v8bf*)(p);                       // K = kbase+koff .. +7
  v8bf hi = *(const v8bf*)(p + 16);                  // K = kbase+16+koff .. +7
  return concat16(lo, hi);
}

// B fragment 32x16 bf16 from transposed weights Wt[n][k] (ld = NW).
// Lanes 0-15: N=lane, K {0..7,16..23}; lanes 16-31: N=lane-16, K {8..15,24..31}.
__device__ __forceinline__ v16bf load_b_frag(const bf16_t* Wt, int n0, int lane, int kbase) {
  const int n      = n0 + (lane & 15);
  const int kstart = kbase + ((lane >> 4) << 3);     // +0 or +8
  const bf16_t* p = Wt + n * NW + kstart;
  v8bf lo = *(const v8bf*)(p);
  v8bf hi = *(const v8bf*)(p + 16);
  return concat16(lo, hi);
}

template<bool FULL>
__global__ __launch_bounds__(256)
void tpmlp_fused(const float* __restrict__ emb, const float* __restrict__ x1,
                 const float* __restrict__ x2,
                 const float* __restrict__ w0, const float* __restrict__ w1,
                 const float* __restrict__ w2,
                 const float* __restrict__ p000, const float* __restrict__ p011,
                 const float* __restrict__ p102, const float* __restrict__ p113,
                 const float* __restrict__ p114,
                 float* __restrict__ out, int rowOffset, int Etot)
{
  __shared__ __align__(16) bf16_t lds[LDS_ELEMS];
  bf16_t* W0t = lds + OFF_W0T;
  bf16_t* W1t = lds + OFF_W1T;
  bf16_t* W2t = lds + OFF_W2T;
  bf16_t* Pt  = lds + OFF_PT;
  bf16_t* h   = lds + OFF_H;

  const int tid     = threadIdx.x;
  const int rowBase = rowOffset + blockIdx.x * MTILE;

  // Warm L2/WGP$ with this block's x1 tile while the MLP runs (epilogue reads it late).
  {
    const float* base = x1 + (size_t)rowBase * X1S;
#pragma unroll
    for (int ofs = 0; ofs < MTILE * X1S; ofs += 256 * 32)
      __builtin_prefetch(base + ofs + tid * 32, 0, 1);
  }

  const float invS8 = 0.35355339059327373f;  // 1/sqrt(8), folded into W0
  const float inv64 = 0.125f;                // 1/sqrt(64), folded into W1/W2/P

  // Cooperative: load all weights into LDS, transposed + prescaled, f32 -> bf16.
  for (int i = tid; i < D_EMB * NW; i += 256) {
    int r = i >> 6, c = i & 63;
    W0t[c * D_EMB + r] = (bf16_t)(w0[i] * invS8);
  }
  for (int i = tid; i < NW * NW; i += 256) {
    int r = i >> 6, c = i & 63;
    W1t[c * NW + r] = (bf16_t)(w1[i] * inv64);
    W2t[c * NW + r] = (bf16_t)(w2[i] * inv64);
  }
  for (int i = tid; i < NW * MUL0; i += 256) {
    int r = i >> 7, c = i & 127;
    Pt[(c      ) * NW + r] = (bf16_t)(p000[i] * inv64);
    Pt[(c + 128) * NW + r] = (bf16_t)(p011[i] * inv64);
  }
  for (int i = tid; i < NW * MUL1; i += 256) {
    int r = i >> 6, c = i & 63;
    Pt[(c + 256) * NW + r] = (bf16_t)(p102[i] * inv64);
    Pt[(c + 320) * NW + r] = (bf16_t)(p113[i] * inv64);
    Pt[(c + 384) * NW + r] = (bf16_t)(p114[i] * inv64);
  }

  const int lane   = tid & 31;
  const int wave   = tid >> 5;
  const int m0     = wave << 4;           // 16-row strip within the 128-row tile
  const int rlocal = (lane >> 4) << 3;    // D-layout: lanes>=16 cover rows m0+8..+15
  const int cIn    = lane & 15;           // D-layout: column within tile

  // ---- Layer 0 A fragment straight from global emb (K=8, zero-padded to 32) ----
  v16bf a0;
#pragma unroll
  for (int j = 0; j < 16; ++j) a0[j] = (bf16_t)0.0f;
  {
    const int row = rowBase + m0 + (lane & 15);
    if (lane < 16 && (FULL || row < Etot)) {
      const float4* ep = (const float4*)(emb + (size_t)row * D_EMB);
      float4 e0 = ep[0], e1 = ep[1];
      a0[0] = (bf16_t)e0.x; a0[1] = (bf16_t)e0.y; a0[2] = (bf16_t)e0.z; a0[3] = (bf16_t)e0.w;
      a0[4] = (bf16_t)e1.x; a0[5] = (bf16_t)e1.y; a0[6] = (bf16_t)e1.z; a0[7] = (bf16_t)e1.w;
    }
  }

  __syncthreads();  // weights visible

  // ---- Layer 0: (16x8) @ (8x64), K padded to 32 ----
#pragma unroll
  for (int t = 0; t < 4; ++t) {
    v16bf b;
#pragma unroll
    for (int j = 0; j < 16; ++j) b[j] = (bf16_t)0.0f;
    if (lane < 16) {                          // only K=0..7 nonzero
      v8bf v = *(const v8bf*)(W0t + (t * 16 + lane) * D_EMB);
#pragma unroll
      for (int j = 0; j < 8; ++j) b[j] = v[j];
    }
    v8f acc = {0.f,0.f,0.f,0.f,0.f,0.f,0.f,0.f};
    acc = wmma_bf16(a0, b, acc);
    const int col = t * 16 + cIn;
#pragma unroll
    for (int r = 0; r < 8; ++r)
      h[(m0 + rlocal + r) * NW + col] = (bf16_t)gelu_fast(acc[r]);
  }
  __syncthreads();

  // ---- Layers 1 & 2: (16x64) @ (64x64) ----
  auto run_layer = [&](const bf16_t* Wt) {
    v16bf af0 = load_a_frag(h, m0, lane, 0);
    v16bf af1 = load_a_frag(h, m0, lane, 32);
    v8f accs[4];
#pragma unroll
    for (int t = 0; t < 4; ++t) {
      v8f acc = {0.f,0.f,0.f,0.f,0.f,0.f,0.f,0.f};
      acc = wmma_bf16(af0, load_b_frag(Wt, t * 16, lane, 0),  acc);
      acc = wmma_bf16(af1, load_b_frag(Wt, t * 16, lane, 32), acc);
      accs[t] = acc;
    }
#pragma unroll
    for (int t = 0; t < 4; ++t) {
      const int col = t * 16 + cIn;
#pragma unroll
      for (int r = 0; r < 8; ++r)
        h[(m0 + rlocal + r) * NW + col] = (bf16_t)gelu_fast(accs[t][r]);
    }
    __syncthreads();
  };
  run_layer(W1t);
  run_layer(W2t);

  // ---- Projections (16x64)@(64x448) fused with tensor-product epilogue ----
  const v16bf wf0 = load_a_frag(h, m0, lane, 0);
  const v16bf wf1 = load_a_frag(h, m0, lane, 32);

  // Per-lane row data / pointers for its 8 rows (reused across all 28 tiles)
  float s2v[8], v2x[8], v2y[8], v2z[8];
  const float* x1row[8];
  float*       outrow[8];
  int          rowv[8];
#pragma unroll
  for (int r = 0; r < 8; ++r) {
    const int row = rowBase + m0 + rlocal + r;
    rowv[r]   = row;
    x1row[r]  = x1 + (size_t)row * X1S;
    outrow[r] = out + (size_t)row * OUTS;
    float4 q = {0.f, 0.f, 0.f, 0.f};
    if (FULL || row < Etot) q = *(const float4*)(x2 + (size_t)row * 4);
    s2v[r] = q.x; v2x[r] = q.y; v2y[r] = q.z; v2z[r] = q.w;
  }

  const float C110 = 0.5773502691896258f;  // 1/sqrt(3)
  const float C111 = 0.7071067811865476f;  // 1/sqrt(2)

#pragma unroll
  for (int t = 0; t < 28; ++t) {
    v8f acc = {0.f,0.f,0.f,0.f,0.f,0.f,0.f,0.f};
    acc = wmma_bf16(wf0, load_b_frag(Pt, t * 16, lane, 0),  acc);
    acc = wmma_bf16(wf1, load_b_frag(Pt, t * 16, lane, 32), acc);
    const int col = t * 16 + cIn;   // 0..447; range is tile-uniform (boundaries %16==0)

    if (col < 128) {                           // out0 = w000 * s1 * s2
      const int u = col;
#pragma unroll
      for (int r = 0; r < 8; ++r) {
        if (FULL || rowv[r] < Etot) {
          float s1 = x1row[r][u];
          outrow[r][u] = acc[r] * s1 * s2v[r];
        }
      }
    } else if (col < 256) {                    // out1 = (w011*s1) outer v2
      const int u = col - 128;
#pragma unroll
      for (int r = 0; r < 8; ++r) {
        if (FULL || rowv[r] < Etot) {
          float g = acc[r] * x1row[r][u];
          float* o = outrow[r] + 128 + 3 * u;
          o[0] = g * v2x[r]; o[1] = g * v2y[r]; o[2] = g * v2z[r];
        }
      }
    } else if (col < 320) {                    // out2 = (w102*s2) * v1
      const int u = col - 256;
#pragma unroll
      for (int r = 0; r < 8; ++r) {
        if (FULL || rowv[r] < Etot) {
          const float* vp = x1row[r] + MUL0 + 3 * u;
          float g = acc[r] * s2v[r];
          float* o = outrow[r] + 512 + 3 * u;
          o[0] = g * vp[0]; o[1] = g * vp[1]; o[2] = g * vp[2];
        }
      }
    } else if (col < 384) {                    // out3 = w113 * <v1,v2> * C110
      const int u = col - 320;
#pragma unroll
      for (int r = 0; r < 8; ++r) {
        if (FULL || rowv[r] < Etot) {
          const float* vp = x1row[r] + MUL0 + 3 * u;
          float d = vp[0] * v2x[r] + vp[1] * v2y[r] + vp[2] * v2z[r];
          outrow[r][704 + u] = acc[r] * d * C110;
        }
      }
    } else {                                   // out4 = w114 * (v1 x v2) * C111
      const int u = col - 384;
#pragma unroll
      for (int r = 0; r < 8; ++r) {
        if (FULL || rowv[r] < Etot) {
          const float* vp = x1row[r] + MUL0 + 3 * u;
          float g = acc[r] * C111;
          float* o = outrow[r] + 768 + 3 * u;
          o[0] = g * (vp[1] * v2z[r] - vp[2] * v2y[r]);
          o[1] = g * (vp[2] * v2x[r] - vp[0] * v2z[r]);
          o[2] = g * (vp[0] * v2y[r] - vp[1] * v2x[r]);
        }
      }
    }
  }
}

extern "C" void kernel_launch(void* const* d_in, const int* in_sizes, int n_in,
                              void* d_out, int out_size, void* d_ws, size_t ws_size,
                              hipStream_t stream) {
  const float* emb  = (const float*)d_in[0];
  const float* x1   = (const float*)d_in[1];
  const float* x2   = (const float*)d_in[2];
  const float* w0   = (const float*)d_in[3];
  const float* w1   = (const float*)d_in[4];
  const float* w2   = (const float*)d_in[5];
  const float* p000 = (const float*)d_in[6];
  const float* p011 = (const float*)d_in[7];
  const float* p102 = (const float*)d_in[8];
  const float* p113 = (const float*)d_in[9];
  const float* p114 = (const float*)d_in[10];

  const int Etot       = in_sizes[0] / D_EMB;   // 131072
  const int fullBlocks = Etot / MTILE;          // 1024 (divides exactly)
  const int rem        = Etot % MTILE;

  if (fullBlocks)
    tpmlp_fused<true><<<fullBlocks, 256, 0, stream>>>(emb, x1, x2, w0, w1, w2,
                                                      p000, p011, p102, p113, p114,
                                                      (float*)d_out, 0, Etot);
  if (rem)
    tpmlp_fused<false><<<1, 256, 0, stream>>>(emb, x1, x2, w0, w1, w2,
                                              p000, p011, p102, p113, p114,
                                              (float*)d_out, fullBlocks * MTILE, Etot);
}
